// model_42125039239251
// MI455X (gfx1250) — compile-verified, matrix operand order checked by reading, not run on hardware
//
#include <hip/hip_runtime.h>

// ---------------------------------------------------------------------------
// HGNN attention layer for MI455X (gfx1250, wave32, WMMA bf16).
// Pipeline:
//   K0  swizzle 4 weight matrices (f32 -> bf16, WMMA B-fragment order)
//   K1  per-edge gather + dual GEMM (x_K, x_V) via v_wmma_f32_16x16x32_bf16.
//       Two 16-edge A-tiles per wave so every B-fragment load feeds 2 WMMAs
//       (halves L2 fragment traffic vs. 1 tile/wave). Attention logits +
//       leaky-relu + segment-max (atomic, order-preserving uint encoding).
//   K2  exp(alpha - max) + segment-sum
//   K3  norm reweight + second segment-sum
//   K4  weighted scatter-add of x_V into per-dst accumulator (atomic f32)
//   K5a +2*att_r, LayerNorm0
//   K5b 2-layer MLP via WMMA (relu between), residual pre-sum
//   K5c LayerNorm1 -> d_out
// ---------------------------------------------------------------------------

#define N_SRC 50000
#define N_DSTN 50000
#define EDGES 800000
#define D_IN 64
#define HID 128
#define NHEAD 4
#define CDIM 32

typedef __attribute__((ext_vector_type(16))) __bf16 v16bf;
typedef __attribute__((ext_vector_type(8)))  float  v8f;

__device__ __forceinline__ __bf16 f2bf(float f) {
    unsigned u = __float_as_uint(f);
    unsigned r = (u + 0x7FFFu + ((u >> 16) & 1u)) >> 16;   // round-nearest-even
    union { unsigned short s; __bf16 b; } cv;
    cv.s = (unsigned short)r;
    return cv.b;
}

// K index inside a 16x32 bf16 A-fragment (and mirrored B-fragment):
// lanes 0-15 hold K {0..7,16..23}, lanes 16-31 hold K {8..15,24..31}.
__device__ __forceinline__ int klocal(int p, int hi) {
    return (p & 7) + ((p >> 3) << 4) + (hi << 3);
}

// Order-preserving float <-> uint for atomic max
__device__ __forceinline__ unsigned encF(float f) {
    unsigned u = __float_as_uint(f);
    return (u & 0x80000000u) ? ~u : (u | 0x80000000u);
}
__device__ __forceinline__ float decF(unsigned u) {
    unsigned v = (u & 0x80000000u) ? (u & 0x7FFFFFFFu) : ~u;
    return __uint_as_float(v);
}

// ---------------------------------------------------------------------------
// K0: pre-swizzle a [128,128] f32 weight (used as B = W^T) into bf16
// fragment order: dst[((kt*8+nt)*32 + lane)*16 + p] = W[n, k]
// ---------------------------------------------------------------------------
__global__ __launch_bounds__(256) void k_swz(const float* __restrict__ W,
                                             __bf16* __restrict__ dstw) {
    int t = blockIdx.x * 256 + threadIdx.x;
    if (t >= HID * HID) return;
    int p = t & 15;
    int lane = (t >> 4) & 31;
    int tile = t >> 9;            // kt*8 + nt
    int kt = tile >> 3, nt = tile & 7;
    int hi = lane >> 4;
    int n = nt * 16 + (lane & 15);
    int k = kt * 32 + klocal(p, hi);
    dstw[t] = f2bf(W[n * HID + k]);
}

// ---------------------------------------------------------------------------
// K1: edge GEMM + attention logits. One wave per 32-edge tile (2 A-tiles).
// ---------------------------------------------------------------------------
__global__ __launch_bounds__(32) void k_edge_gemm(
    const float* __restrict__ src, const float* __restrict__ dstF,
    const int* __restrict__ eidx,                  // [2, E]
    const float* __restrict__ b_enc, const float* __restrict__ b_enck,
    const float* __restrict__ att_r,               // [128]
    const __bf16* __restrict__ wK, const __bf16* __restrict__ wV,
    float* __restrict__ xV,                        // [E,128]
    float* __restrict__ alphaBuf,                  // [E,4]
    unsigned* __restrict__ aMaxEnc)                // [N,4]
{
    __shared__ __bf16 aT[32][HID];
    __shared__ float alphaAcc[32][NHEAD];

    const int lane = threadIdx.x;
    const int base = blockIdx.x * 32;
    if (base >= EDGES) return;

    // stage 32 edges of concat(src_row, dst_row) as bf16
    for (int e = 0; e < 32; ++e) {
        int edge = base + e;
        int s = eidx[edge];
        int d = eidx[EDGES + edge];
        aT[e][lane]        = f2bf(src[s * D_IN + lane]);
        aT[e][lane + 32]   = f2bf(src[s * D_IN + 32 + lane]);
        aT[e][lane + 64]   = f2bf(dstF[d * D_IN + lane]);
        aT[e][lane + 96]   = f2bf(dstF[d * D_IN + 32 + lane]);
    }
    {
        float* aa = &alphaAcc[0][0];
        aa[lane] = 0.f; aa[lane + 32] = 0.f;
        aa[lane + 64] = 0.f; aa[lane + 96] = 0.f;
    }
    __syncthreads();

    const int m = lane & 15, hi = lane >> 4;
    v16bf af0[4], af1[4];
    for (int kt = 0; kt < 4; ++kt)
        for (int p = 0; p < 16; ++p) {
            const int kk = kt * 32 + klocal(p, hi);
            af0[kt][p] = aT[m][kk];
            af1[kt][p] = aT[16 + m][kk];
        }

    for (int nt = 0; nt < 8; ++nt) {
        v8f accK0 = {}; v8f accV0 = {};
        v8f accK1 = {}; v8f accV1 = {};
        for (int kt = 0; kt < 4; ++kt) {
            const v16bf bK = *(const v16bf*)(wK + ((size_t)((kt * 8 + nt) * 32 + lane)) * 16);
            const v16bf bV = *(const v16bf*)(wV + ((size_t)((kt * 8 + nt) * 32 + lane)) * 16);
            accK0 = __builtin_amdgcn_wmma_f32_16x16x32_bf16(false, af0[kt], false, bK,
                                                            (short)0, accK0, false, false);
            accV0 = __builtin_amdgcn_wmma_f32_16x16x32_bf16(false, af0[kt], false, bV,
                                                            (short)0, accV0, false, false);
            accK1 = __builtin_amdgcn_wmma_f32_16x16x32_bf16(false, af1[kt], false, bK,
                                                            (short)0, accK1, false, false);
            accV1 = __builtin_amdgcn_wmma_f32_16x16x32_bf16(false, af1[kt], false, bV,
                                                            (short)0, accV1, false, false);
        }
        const int j = nt * 16 + m;          // output column 0..127
        const float bk = b_enck[j], bv = b_enc[j], at = att_r[j];
        const int h = nt >> 1;              // one head spans 2 n-tiles
        for (int r = 0; r < 8; ++r) {
            const int row = r + 8 * hi;     // C-layout: lanes>=16 hold M=r+8
            // tile 0 (edges base .. base+15)
            xV[(size_t)(base + row) * HID + j] = accV0[r] + bv;
            float kv0 = (accK0[r] + bk) * at;
            kv0 += __shfl_xor(kv0, 1, 32);
            kv0 += __shfl_xor(kv0, 2, 32);
            kv0 += __shfl_xor(kv0, 4, 32);
            kv0 += __shfl_xor(kv0, 8, 32);
            if (m == 0) alphaAcc[row][h] += kv0;
            // tile 1 (edges base+16 .. base+31)
            xV[(size_t)(base + 16 + row) * HID + j] = accV1[r] + bv;
            float kv1 = (accK1[r] + bk) * at;
            kv1 += __shfl_xor(kv1, 1, 32);
            kv1 += __shfl_xor(kv1, 2, 32);
            kv1 += __shfl_xor(kv1, 4, 32);
            kv1 += __shfl_xor(kv1, 8, 32);
            if (m == 0) alphaAcc[16 + row][h] += kv1;
        }
    }
    __syncthreads();

    // leaky-relu, store logits, segment max (32 edges x 4 heads)
    const float* aa = &alphaAcc[0][0];
    for (int t = lane; t < 32 * NHEAD; t += 32) {
        int e = t >> 2, h = t & 3;
        float a = aa[t];
        a = a > 0.f ? a : 0.2f * a;
        int edge = base + e;
        alphaBuf[(size_t)edge * NHEAD + h] = a;
        int d = eidx[EDGES + edge];
        atomicMax(&aMaxEnc[(size_t)d * NHEAD + h], encF(a));
    }
}

// ---------------------------------------------------------------------------
// K2: exp(alpha - max) + segment sum
// ---------------------------------------------------------------------------
__global__ __launch_bounds__(256) void k_softmax1(
    const int* __restrict__ eidx, const unsigned* __restrict__ aMaxEnc,
    float* __restrict__ alpha, float* __restrict__ aSum) {
    int t = blockIdx.x * 256 + threadIdx.x;
    if (t >= EDGES * NHEAD) return;
    int e = t >> 2, h = t & 3;
    int d = eidx[EDGES + e];
    float mx = decF(aMaxEnc[(size_t)d * NHEAD + h]);
    float ex = __expf(alpha[t] - mx);
    alpha[t] = ex;
    atomicAdd(&aSum[(size_t)d * NHEAD + h], ex);
}

// ---------------------------------------------------------------------------
// K3: normalize, reweight by norm, second segment sum
// ---------------------------------------------------------------------------
__global__ __launch_bounds__(256) void k_softmax2(
    const int* __restrict__ eidx, const float* __restrict__ norm,
    const float* __restrict__ aSum, float* __restrict__ alpha,
    float* __restrict__ outSum) {
    int t = blockIdx.x * 256 + threadIdx.x;
    if (t >= EDGES * NHEAD) return;
    int e = t >> 2, h = t & 3;
    int d = eidx[EDGES + e];
    float w = norm[e] * (alpha[t] / (aSum[(size_t)d * NHEAD + h] + 1e-16f)) + 1e-6f;
    alpha[t] = w;
    atomicAdd(&outSum[(size_t)d * NHEAD + h], w);
}

// ---------------------------------------------------------------------------
// K4: weighted scatter-add of x_V into per-destination accumulator
// ---------------------------------------------------------------------------
__global__ __launch_bounds__(256) void k_scatter(
    const int* __restrict__ eidx, const float* __restrict__ alpha,
    const float* __restrict__ outSum, const float* __restrict__ xV,
    float* __restrict__ outAcc) {
    int t = blockIdx.x * 256 + threadIdx.x;
    if (t >= EDGES * HID) return;
    int e = t >> 7, j = t & 127, h = j >> 5;
    int d = eidx[EDGES + e];
    float a = alpha[(size_t)e * NHEAD + h] / (outSum[(size_t)d * NHEAD + h] + 1e-16f);
    atomicAdd(&outAcc[(size_t)d * HID + j], xV[(size_t)t] * a);
}

// ---------------------------------------------------------------------------
// K5a/K5c: LayerNorm over 128 cols, one wave per row (8 rows / 256-thr block).
// Optionally adds addscale*addv[col] (broadcast) before normalizing.
// ---------------------------------------------------------------------------
__global__ __launch_bounds__(256) void k_ln(
    const float* __restrict__ x, const float* __restrict__ addv, float addscale,
    const float* __restrict__ g, const float* __restrict__ b,
    float* __restrict__ out, int nrows) {
    int row = blockIdx.x * 8 + (threadIdx.x >> 5);
    if (row >= nrows) return;
    int lane = threadIdx.x & 31;
    float v[4]; float s = 0.f, s2 = 0.f;
    for (int q = 0; q < 4; ++q) {
        int c = lane + 32 * q;
        float t = x[(size_t)row * HID + c];
        if (addv) t += addscale * addv[c];
        v[q] = t; s += t; s2 += t * t;
    }
    for (int msk = 1; msk < 32; msk <<= 1) {
        s  += __shfl_xor(s,  msk, 32);
        s2 += __shfl_xor(s2, msk, 32);
    }
    float mu  = s * (1.f / 128.f);
    float var = s2 * (1.f / 128.f) - mu * mu;
    float rs  = rsqrtf(var + 1e-5f);
    for (int q = 0; q < 4; ++q) {
        int c = lane + 32 * q;
        out[(size_t)row * HID + c] = (v[q] - mu) * rs * g[c] + b[c];
    }
}

// ---------------------------------------------------------------------------
// K5b: 2-layer MLP (128->128->128, relu between) via WMMA; writes h0+relu(dec)
// One wave per 16-row tile.
// ---------------------------------------------------------------------------
__global__ __launch_bounds__(32) void k_mlp(
    const float* __restrict__ h0, const __bf16* __restrict__ w1s,
    const __bf16* __restrict__ w2s, const float* __restrict__ b1,
    const float* __restrict__ b2, float* __restrict__ zPre, int nrows) {
    __shared__ __bf16 yT[16][HID];
    const int lane = threadIdx.x;
    const int base = blockIdx.x * 16;
    if (base >= nrows) return;
    const int m = lane & 15, hi = lane >> 4;

    v16bf af[4];
    for (int kt = 0; kt < 4; ++kt)
        for (int p = 0; p < 16; ++p)
            af[kt][p] = f2bf(h0[(size_t)(base + m) * HID + kt * 32 + klocal(p, hi)]);

    // GEMM1: y = relu(h0 @ W1^T + b1)  -> LDS (bf16)
    for (int nt = 0; nt < 8; ++nt) {
        v8f acc = {};
        for (int kt = 0; kt < 4; ++kt) {
            const v16bf bfr = *(const v16bf*)(w1s + ((size_t)((kt * 8 + nt) * 32 + lane)) * 16);
            acc = __builtin_amdgcn_wmma_f32_16x16x32_bf16(false, af[kt], false, bfr,
                                                          (short)0, acc, false, false);
        }
        int j = nt * 16 + m; float bb = b1[j];
        for (int r = 0; r < 8; ++r) {
            float v = acc[r] + bb;
            yT[r + 8 * hi][j] = f2bf(v > 0.f ? v : 0.f);
        }
    }
    __syncthreads();

    for (int kt = 0; kt < 4; ++kt)
        for (int p = 0; p < 16; ++p)
            af[kt][p] = yT[m][kt * 32 + klocal(p, hi)];

    // GEMM2: dec = y @ W2^T + b2 ; zPre = h0 + relu(dec)
    for (int nt = 0; nt < 8; ++nt) {
        v8f acc = {};
        for (int kt = 0; kt < 4; ++kt) {
            const v16bf bfr = *(const v16bf*)(w2s + ((size_t)((kt * 8 + nt) * 32 + lane)) * 16);
            acc = __builtin_amdgcn_wmma_f32_16x16x32_bf16(false, af[kt], false, bfr,
                                                          (short)0, acc, false, false);
        }
        int j = nt * 16 + m; float bb = b2[j];
        for (int r = 0; r < 8; ++r) {
            int row = base + r + 8 * hi;
            float d = acc[r] + bb; d = d > 0.f ? d : 0.f;
            zPre[(size_t)row * HID + j] = h0[(size_t)row * HID + j] + d;
        }
    }
}

// ---------------------------------------------------------------------------
extern "C" void kernel_launch(void* const* d_in, const int* in_sizes, int n_in,
                              void* d_out, int out_size, void* d_ws, size_t ws_size,
                              hipStream_t stream) {
    const float* src    = (const float*)d_in[0];
    const float* dstF   = (const float*)d_in[1];
    const int*   eidx   = (const int*)d_in[2];
    const float* norm   = (const float*)d_in[3];
    const float* W_enc  = (const float*)d_in[4];
    const float* b_enc  = (const float*)d_in[5];
    const float* W_enck = (const float*)d_in[6];
    const float* b_enck = (const float*)d_in[7];
    const float* W1     = (const float*)d_in[8];
    const float* b1     = (const float*)d_in[9];
    const float* W2     = (const float*)d_in[10];
    const float* b2     = (const float*)d_in[11];
    const float* att_r  = (const float*)d_in[12];
    const float* ln0g   = (const float*)d_in[13];
    const float* ln0b   = (const float*)d_in[14];
    const float* ln1g   = (const float*)d_in[15];
    const float* ln1b   = (const float*)d_in[16];
    float* out = (float*)d_out;

    char* ws = (char*)d_ws;
    size_t off = 0;
    auto alloc = [&](size_t bytes) -> void* {
        void* p = ws + off;
        off = (off + bytes + 255) & ~(size_t)255;
        return p;
    };

    __bf16* wKs = (__bf16*)alloc((size_t)HID * HID * 2);
    __bf16* wVs = (__bf16*)alloc((size_t)HID * HID * 2);
    __bf16* w1s = (__bf16*)alloc((size_t)HID * HID * 2);
    __bf16* w2s = (__bf16*)alloc((size_t)HID * HID * 2);

    // contiguous zero-init region
    size_t zeroStart = off;
    unsigned* aMaxEnc = (unsigned*)alloc((size_t)N_DSTN * NHEAD * 4);
    float*    aSum    = (float*)   alloc((size_t)N_DSTN * NHEAD * 4);
    float*    outSum  = (float*)   alloc((size_t)N_DSTN * NHEAD * 4);
    float*    outAcc  = (float*)   alloc((size_t)N_DSTN * HID * 4);
    size_t zeroBytes = off - zeroStart;

    float* alphaBuf = (float*)alloc((size_t)EDGES * NHEAD * 4);
    float* xV       = (float*)alloc((size_t)EDGES * HID * 4);
    float* h0       = (float*)alloc((size_t)N_DSTN * HID * 4);
    float* zPre     = (float*)alloc((size_t)N_DSTN * HID * 4);

    // K0: weight swizzle (4 launches)
    k_swz<<<(HID * HID + 255) / 256, 256, 0, stream>>>(W_enck, wKs);
    k_swz<<<(HID * HID + 255) / 256, 256, 0, stream>>>(W_enc,  wVs);
    k_swz<<<(HID * HID + 255) / 256, 256, 0, stream>>>(W1,     w1s);
    k_swz<<<(HID * HID + 255) / 256, 256, 0, stream>>>(W2,     w2s);

    hipMemsetAsync(ws + zeroStart, 0, zeroBytes, stream);

    // K1: edge GEMM + logits + segment max  (E/32 = 25000 waves)
    k_edge_gemm<<<EDGES / 32, 32, 0, stream>>>(src, dstF, eidx, b_enc, b_enck,
                                               att_r, wKs, wVs, xV, alphaBuf, aMaxEnc);

    // K2/K3: two-round segment softmax
    k_softmax1<<<(EDGES * NHEAD + 255) / 256, 256, 0, stream>>>(eidx, aMaxEnc, alphaBuf, aSum);
    k_softmax2<<<(EDGES * NHEAD + 255) / 256, 256, 0, stream>>>(eidx, norm, aSum, alphaBuf, outSum);

    // K4: weighted scatter-aggregate
    k_scatter<<<(EDGES * HID + 255) / 256, 256, 0, stream>>>(eidx, alphaBuf, outSum, xV, outAcc);

    // K5: +2*att_r, LN0, MLP, residual, LN1
    k_ln<<<(N_DSTN + 7) / 8, 256, 0, stream>>>(outAcc, att_r, 2.0f, ln0g, ln0b, h0, N_DSTN);
    k_mlp<<<N_DSTN / 16, 32, 0, stream>>>(h0, w1s, w2s, b1, b2, zPre, N_DSTN);
    k_ln<<<(N_DSTN + 7) / 8, 256, 0, stream>>>(zPre, (const float*)nullptr, 0.0f,
                                               ln1g, ln1b, out, N_DSTN);
}
